// Forecaster_71914932404495
// MI455X (gfx1250) — matrix operand order repaired and checked by reading, not compiled, not run on hardware
//
#include <hip/hip_runtime.h>

// ---------------------------------------------------------------------------
// MI455X (gfx1250, wave32) implementation of the 2-level ConvGRU forecaster.
// Gate projections (the GEMM hotspots) run on v_wmma_f32_16x16x32_f16 with a
// fused GRU epilogue. A/B operands are pre-swizzled into WMMA fragment tile
// order so every fragment is a single aligned 32B vector load, and the K-loop
// is software-pipelined (load kt+1 while WMMAs consume kt).
// ---------------------------------------------------------------------------

#define LEAK 0.2f

typedef __attribute__((ext_vector_type(16))) _Float16 v16h;
typedef __attribute__((ext_vector_type(8)))  float    v8f;

__device__ __forceinline__ float dev_sigmoid(float x) { return 1.0f / (1.0f + __expf(-x)); }
__device__ __forceinline__ float dev_lrelu(float x)   { return x > 0.0f ? x : LEAK * x; }

// ---------------------------------------------------------------------------
// 5x5 conv, pad=2, optional leaky-relu. Supports virtual concat of two inputs
// along channels (in0: C0 channels, in1: C1 channels) to avoid materializing
// concat([x1, h1]).
// ---------------------------------------------------------------------------
__global__ void conv5x5_kernel(const float* __restrict__ in0, int C0,
                               const float* __restrict__ in1, int C1,
                               const float* __restrict__ w,
                               const float* __restrict__ bias,
                               float* __restrict__ out,
                               int B, int CO, int H, int W, int doRelu)
{
    int idx = blockIdx.x * blockDim.x + threadIdx.x;
    int total = B * CO * H * W;
    if (idx >= total) return;
    int x = idx % W;
    int y = (idx / W) % H;
    int o = (idx / (W * H)) % CO;
    int b = idx / (W * H * CO);
    int CI = C0 + C1;
    float acc = bias[o];
    for (int ci = 0; ci < CI; ++ci) {
        const float* src = (ci < C0)
            ? in0 + ((size_t)b * C0 + ci) * H * W
            : in1 + ((size_t)b * C1 + (ci - C0)) * H * W;
        const float* wr = w + ((size_t)o * CI + ci) * 25;
        #pragma unroll
        for (int ky = 0; ky < 5; ++ky) {
            int yy = y + ky - 2;
            if (yy < 0 || yy >= H) continue;
            #pragma unroll
            for (int kx = 0; kx < 5; ++kx) {
                int xx = x + kx - 2;
                if (xx < 0 || xx >= W) continue;
                acc += src[yy * W + xx] * wr[ky * 5 + kx];
            }
        }
    }
    out[idx] = doRelu ? dev_lrelu(acc) : acc;
}

// ---------------------------------------------------------------------------
// 9-link bilinear warp. Writes M directly in WMMA B-fragment tile order:
//   Mf[kt][pt][lane][e]  (halfs), lane = 16*(kk>>4) + (p&15), e = kk&15,
// where k = link*C + c, kk = k&31, kt = k>>5, pt = p>>4.
// Per thread (fixed p) the writes along c are contiguous 16-half (32B) runs.
// ---------------------------------------------------------------------------
__global__ void warp_kernel(const float* __restrict__ h,
                            const float* __restrict__ flow,
                            _Float16* __restrict__ Mf,
                            int B, int C, int H, int W, int PT)
{
    int idx = blockIdx.x * blockDim.x + threadIdx.x;
    int total = B * 9 * H * W;
    if (idx >= total) return;
    int x = idx % W;
    int y = (idx / W) % H;
    int l = (idx / (W * H)) % 9;
    int b = idx / (W * H * 9);
    int HW = H * W;
    float u = flow[((size_t)b * 18 + l) * HW + y * W + x];
    float v = flow[((size_t)b * 18 + 9 + l) * HW + y * W + x];
    float gx = ((float)x * 2.0f / (float)W - 1.0f) + u;
    float gy = ((float)y * 2.0f / (float)H - 1.0f) + v;
    float ix = (gx + 1.0f) * 0.5f * (float)(W - 1);
    float iy = (gy + 1.0f) * 0.5f * (float)(H - 1);
    float x0f = floorf(ix), y0f = floorf(iy);
    float wx = ix - x0f, wy = iy - y0f;
    float W1 = (float)(W - 1), H1 = (float)(H - 1);
    float xs[2]  = { x0f, x0f + 1.0f };
    float ys[2]  = { y0f, y0f + 1.0f };
    float wxs[2] = { 1.0f - wx, wx };
    float wys[2] = { 1.0f - wy, wy };
    int   off[4];
    float wt[4];
    #pragma unroll
    for (int ty = 0; ty < 2; ++ty) {
        #pragma unroll
        for (int tx = 0; tx < 2; ++tx) {
            float xf = xs[tx], yf = ys[ty];
            bool valid = (xf >= 0.0f) && (xf <= W1) && (yf >= 0.0f) && (yf <= H1);
            int xc = (int)fminf(fmaxf(xf, 0.0f), W1);
            int yc = (int)fminf(fmaxf(yf, 0.0f), H1);
            off[ty * 2 + tx] = yc * W + xc;
            wt[ty * 2 + tx]  = wxs[tx] * wys[ty] * (valid ? 1.0f : 0.0f);
        }
    }
    size_t p  = (size_t)b * HW + (size_t)y * W + x;
    int ptile = (int)(p >> 4);
    int ccol  = (int)(p & 15);
    for (int c = 0; c < C; ++c) {
        const float* hb = h + ((size_t)b * C + c) * HW;
        float val = wt[0] * hb[off[0]] + wt[1] * hb[off[1]]
                  + wt[2] * hb[off[2]] + wt[3] * hb[off[3]];
        int k    = l * C + c;
        int kt   = k >> 5;
        int kk   = k & 31;
        int lane = ((kk >> 4) << 4) + ccol;   // 16*half + col
        int e    = kk & 15;
        size_t offm = (((size_t)kt * PT + ptile) * 32 + lane) * 16 + e;
        Mf[offm] = (_Float16)val;
    }
}

// ---------------------------------------------------------------------------
// Convert the three 1x1 gate weights [K][Cout] (f32) into f16 A-matrices in
// WMMA A-fragment tile order: Af[g][ot][kt][lane][e], where lane = 16*half+r,
// k = kt*32 + 16*half + e, o = ot*16 + r.
// ---------------------------------------------------------------------------
__global__ void cvt_gate_weights_kernel(const float* __restrict__ wr,
                                        const float* __restrict__ wu,
                                        const float* __restrict__ wo,
                                        _Float16* __restrict__ Af,
                                        int Cout, int K)
{
    int OT = Cout >> 4;
    int KT = K >> 5;
    int idx = blockIdx.x * blockDim.x + threadIdx.x;
    int total = 3 * OT * KT * 512;
    if (idx >= total) return;
    int e    = idx & 15;
    int lane = (idx >> 4) & 31;
    int tile = idx >> 9;                 // g*OT*KT + ot*KT + kt
    int kt   = tile % KT;
    int ot   = (tile / KT) % OT;
    int g    = tile / (KT * OT);
    int half = lane >> 4;
    int r    = lane & 15;
    int v    = e >> 1;
    int t    = e & 1;
    int k    = kt * 32 + 16 * half + e;
    int o    = ot * 16 + r;
    const float* w = (g == 0) ? wr : ((g == 1) ? wu : wo);
    Af[idx] = (_Float16)w[(size_t)k * Cout + o];
}

// ---------------------------------------------------------------------------
// Fused gate GEMM + GRU epilogue. One wave owns a 16(o) x 16(p) tile for all
// three gates (shared B-tile per K-step, 3 WMMAs per step) and applies
// sigmoid / leaky-relu / blend with the previous hidden state in-register.
// Software-pipelined: fragments for kt+1 are loaded before the kt WMMAs so
// 8 b128 loads stay in flight under each WMMA group.
// Grid is sized so every wave is full (EXEC all-ones at every WMMA).
// ---------------------------------------------------------------------------
template<int KT, int HWSHIFT>
__global__ void gates_wmma_kernel(const _Float16* __restrict__ Mf, // [KT][PT][32][16]
                                  const _Float16* __restrict__ Af, // [3][OT][KT][32][16]
                                  const float* __restrict__ br,
                                  const float* __restrict__ bu,
                                  const float* __restrict__ bo,
                                  const float* __restrict__ hprev,
                                  const float* __restrict__ rx,
                                  const float* __restrict__ ux,
                                  const float* __restrict__ ox,
                                  float* __restrict__ sout,
                                  int Cout, int PT)
{
    int lane = threadIdx.x & 31;
    int wid  = (blockIdx.x * blockDim.x + threadIdx.x) >> 5;
    int OT = Cout >> 4;
    int ot = wid % OT;
    int pt = wid / OT;
    if (pt >= PT) return;               // wave-uniform guard

    const size_t bstride = (size_t)PT * 512;
    const _Float16* bp = Mf + ((size_t)pt * 32 + lane) * 16;
    const _Float16* a0 = Af + (((size_t)(0 * OT + ot) * KT) * 32 + lane) * 16;
    const _Float16* a1 = Af + (((size_t)(1 * OT + ot) * KT) * 32 + lane) * 16;
    const _Float16* a2 = Af + (((size_t)(2 * OT + ot) * KT) * 32 + lane) * 16;

    v8f accR = {0,0,0,0,0,0,0,0};
    v8f accU = {0,0,0,0,0,0,0,0};
    v8f accO = {0,0,0,0,0,0,0,0};

    // Prologue: load fragments for kt = 0.
    v16h bf = *(const v16h*)(bp);
    v16h ar = *(const v16h*)(a0);
    v16h au = *(const v16h*)(a1);
    v16h ao = *(const v16h*)(a2);

    #pragma unroll 2
    for (int kt = 0; kt < KT - 1; ++kt) {
        if (kt + 2 < KT)
            __builtin_prefetch(bp + (size_t)(kt + 2) * bstride, 0, 1);
        // Issue next-step loads before consuming the current fragments.
        v16h bf_n = *(const v16h*)(bp + (size_t)(kt + 1) * bstride);
        v16h ar_n = *(const v16h*)(a0 + (size_t)(kt + 1) * 512);
        v16h au_n = *(const v16h*)(a1 + (size_t)(kt + 1) * 512);
        v16h ao_n = *(const v16h*)(a2 + (size_t)(kt + 1) * 512);
        accR = __builtin_amdgcn_wmma_f32_16x16x32_f16(false, ar, false, bf, (short)0, accR, false, false);
        accU = __builtin_amdgcn_wmma_f32_16x16x32_f16(false, au, false, bf, (short)0, accU, false, false);
        accO = __builtin_amdgcn_wmma_f32_16x16x32_f16(false, ao, false, bf, (short)0, accO, false, false);
        bf = bf_n; ar = ar_n; au = au_n; ao = ao_n;
    }
    // Drain last step.
    accR = __builtin_amdgcn_wmma_f32_16x16x32_f16(false, ar, false, bf, (short)0, accR, false, false);
    accU = __builtin_amdgcn_wmma_f32_16x16x32_f16(false, au, false, bf, (short)0, accU, false, false);
    accO = __builtin_amdgcn_wmma_f32_16x16x32_f16(false, ao, false, bf, (short)0, accO, false, false);

    // Epilogue: D VGPR r -> row ot*16 + 8*half + r, col p = pt*16 + (lane&15).
    int half = lane >> 4;
    int lr   = lane & 15;
    int p    = pt * 16 + lr;
    int b    = p >> HWSHIFT;
    int sp   = p & ((1 << HWSHIFT) - 1);
    #pragma unroll
    for (int r = 0; r < 8; ++r) {
        int o = ot * 16 + 8 * half + r;
        size_t gi = ((size_t)b * Cout + o) << HWSHIFT;
        gi += sp;
        float rv = accR[r] + br[o];
        float uv = accU[r] + bu[o];
        float ov = accO[r] + bo[o];
        float outv;
        if (rx) {                       // cell with input x: add x-path gates
            rv += rx[gi];
            uv += ux[gi];
            float reset = dev_sigmoid(rv);
            outv = dev_lrelu(ox[gi] + reset * ov);
        } else {                        // first cell (x == None)
            float reset = dev_sigmoid(rv);
            outv = dev_lrelu(ov * reset);
        }
        float update = dev_sigmoid(uv);
        sout[gi] = outv * (1.0f - update) + update * hprev[gi];
    }
}

// ---------------------------------------------------------------------------
// Generic transposed conv:  y[b,o,Y,X] = bias[o] + sum x[b,i,h,w] * w[i,o,kh,kw]
// over (h,w) with Y = h*S - PAD + kh, X = w*S - PAD + kw. Matches the JAX
// _deconv (flip + transpose + lhs_dilation).
// ---------------------------------------------------------------------------
__global__ void deconv_kernel(const float* __restrict__ x,
                              const float* __restrict__ w,
                              const float* __restrict__ bias,
                              float* __restrict__ y,
                              int B, int CI, int CO,
                              int Hin, int Win, int Hout, int Wout,
                              int Kk, int S, int PAD)
{
    int idx = blockIdx.x * blockDim.x + threadIdx.x;
    int total = B * CO * Hout * Wout;
    if (idx >= total) return;
    int X = idx % Wout;
    int Y = (idx / Wout) % Hout;
    int o = (idx / (Wout * Hout)) % CO;
    int b = idx / (Wout * Hout * CO);
    float acc = bias[o];
    for (int kh = 0; kh < Kk; ++kh) {
        int ty = Y + PAD - kh;
        if (ty < 0 || (ty % S) != 0) continue;
        int hh = ty / S;
        if (hh >= Hin) continue;
        for (int kw = 0; kw < Kk; ++kw) {
            int tx = X + PAD - kw;
            if (tx < 0 || (tx % S) != 0) continue;
            int ww = tx / S;
            if (ww >= Win) continue;
            const float* xb = x + (size_t)b * CI * Hin * Win + (size_t)hh * Win + ww;
            float s = 0.0f;
            for (int i = 0; i < CI; ++i) {
                s += xb[(size_t)i * Hin * Win]
                   * w[(((size_t)i * CO + o) * Kk + kh) * Kk + kw];
            }
            acc += s;
        }
    }
    y[idx] = acc;
}

// ---------------------------------------------------------------------------
// Orchestration
// ---------------------------------------------------------------------------
extern "C" void kernel_launch(void* const* d_in, const int* in_sizes, int n_in,
                              void* d_out, int out_size, void* d_ws, size_t ws_size,
                              hipStream_t stream)
{
    (void)in_sizes; (void)n_in; (void)out_size; (void)ws_size;

    const float* h0      = (const float*)d_in[0];
    const float* h1      = (const float*)d_in[1];
    const float* fw_w1_0 = (const float*)d_in[2];
    const float* fw_b1_0 = (const float*)d_in[3];
    const float* fw_w2_0 = (const float*)d_in[4];
    const float* fw_b2_0 = (const float*)d_in[5];
    const float* rgw_w_0 = (const float*)d_in[6];
    const float* rgw_b_0 = (const float*)d_in[7];
    const float* ugw_w_0 = (const float*)d_in[8];
    const float* ugw_b_0 = (const float*)d_in[9];
    const float* ogw_w_0 = (const float*)d_in[10];
    const float* ogw_b_0 = (const float*)d_in[11];
    const float* up_w_0  = (const float*)d_in[12];
    const float* up_b_0  = (const float*)d_in[13];
    const float* fw_w1_1 = (const float*)d_in[14];
    const float* fw_b1_1 = (const float*)d_in[15];
    const float* fw_w2_1 = (const float*)d_in[16];
    const float* fw_b2_1 = (const float*)d_in[17];
    const float* rg_w_1  = (const float*)d_in[18];
    const float* rg_b_1  = (const float*)d_in[19];
    const float* ug_w_1  = (const float*)d_in[20];
    const float* ug_b_1  = (const float*)d_in[21];
    const float* og_w_1  = (const float*)d_in[22];
    const float* og_b_1  = (const float*)d_in[23];
    const float* rgw_w_1 = (const float*)d_in[24];
    const float* rgw_b_1 = (const float*)d_in[25];
    const float* ugw_w_1 = (const float*)d_in[26];
    const float* ugw_b_1 = (const float*)d_in[27];
    const float* ogw_w_1 = (const float*)d_in[28];
    const float* ogw_b_1 = (const float*)d_in[29];
    const float* up_w_1  = (const float*)d_in[30];
    const float* up_b_1  = (const float*)d_in[31];
    const float* out_w   = (const float*)d_in[32];
    const float* out_b   = (const float*)d_in[33];

    // Workspace layout (lifetime-based aliasing; needs ~96 MB):
    //   [0,17MB)   : d0, flow0, M0   -> then d1, flow1 -> then s1
    //   [17,55MB)  : M1              -> then x2
    //   [55,57MB)  : Af0, Af1 (f16 gate weights, persistent)
    //   [57,61MB)  : s0
    //   [61,70MB)  : x1
    //   [70,97MB)  : rx, ux, ox
    char* ws = (char*)d_ws;
    const size_t MB = 1ull << 20;
    float*    d0    = (float*)   (ws + 0 * MB);   // (2,32,64,64)    1.0 MB
    float*    flow0 = (float*)   (ws + 1 * MB);   // (2,18,64,64)    0.6 MB
    _Float16* M0    = (_Float16*)(ws + 2 * MB);   // 864*8192 halfs 13.5 MB
    float*    d1    = (float*)   (ws + 0 * MB);   // (2,32,128,128)  4.0 MB (reuse)
    float*    flow1 = (float*)   (ws + 4 * MB);   // (2,18,128,128)  2.3 MB (reuse)
    float*    s1    = (float*)   (ws + 0 * MB);   // (2,64,128,128)  8.0 MB (reuse)
    _Float16* M1    = (_Float16*)(ws + 17 * MB);  // 576*32768 halfs 36.0 MB
    float*    x2    = (float*)   (ws + 17 * MB);  // (2,32,256,256) 16.0 MB (reuse)
    _Float16* Af0   = (_Float16*)(ws + 55 * MB);  // 3*96*864 f16
    _Float16* Af1   = (_Float16*)(ws + 56 * MB);  // 3*64*576 f16
    float*    s0    = (float*)   (ws + 57 * MB);  // (2,96,64,64)    3.0 MB
    float*    x1    = (float*)   (ws + 61 * MB);  // (2,64,128,128)  8.0 MB
    float*    rxb   = (float*)   (ws + 70 * MB);  // (2,64,128,128)  8.0 MB
    float*    uxb   = (float*)   (ws + 79 * MB);
    float*    oxb   = (float*)   (ws + 88 * MB);

    const int TB = 256;
    auto nb = [](long long n) { return (int)((n + 255) / 256); };

    // Gate weight conversion into WMMA A-fragment tile layout
    cvt_gate_weights_kernel<<<nb(3LL * 96 * 864), TB, 0, stream>>>(rgw_w_0, ugw_w_0, ogw_w_0, Af0, 96, 864);
    cvt_gate_weights_kernel<<<nb(3LL * 64 * 576), TB, 0, stream>>>(rgw_w_1, ugw_w_1, ogw_w_1, Af1, 64, 576);

    // ---- cell 0 (x = None, h0: 2x96x64x64) ----
    conv5x5_kernel<<<nb(2LL * 32 * 64 * 64), TB, 0, stream>>>(h0, 96, nullptr, 0, fw_w1_0, fw_b1_0, d0, 2, 32, 64, 64, 1);
    conv5x5_kernel<<<nb(2LL * 18 * 64 * 64), TB, 0, stream>>>(d0, 32, nullptr, 0, fw_w2_0, fw_b2_0, flow0, 2, 18, 64, 64, 0);
    warp_kernel<<<nb(2LL * 9 * 64 * 64), TB, 0, stream>>>(h0, flow0, M0, 2, 96, 64, 64, /*PT=*/512);
    // waves = (96/16) * (8192/16) = 3072 -> 384 blocks of 8 full waves
    gates_wmma_kernel<27, 12><<<384, TB, 0, stream>>>(M0, Af0, rgw_b_0, ugw_b_0, ogw_b_0,
                                                      h0, nullptr, nullptr, nullptr, s0,
                                                      96, /*PT=*/512);
    // upsample: s0 (96,64,64) -> x1 (64,128,128)
    deconv_kernel<<<nb(2LL * 64 * 128 * 128), TB, 0, stream>>>(s0, up_w_0, up_b_0, x1,
                                                               2, 96, 64, 64, 64, 128, 128, 4, 2, 1);

    // ---- cell 1 (x = x1, h1: 2x64x128x128) ----
    conv5x5_kernel<<<nb(2LL * 32 * 128 * 128), TB, 0, stream>>>(x1, 64, h1, 64, fw_w1_1, fw_b1_1, d1, 2, 32, 128, 128, 1);
    conv5x5_kernel<<<nb(2LL * 18 * 128 * 128), TB, 0, stream>>>(d1, 32, nullptr, 0, fw_w2_1, fw_b2_1, flow1, 2, 18, 128, 128, 0);
    warp_kernel<<<nb(2LL * 9 * 128 * 128), TB, 0, stream>>>(h1, flow1, M1, 2, 64, 128, 128, /*PT=*/2048);
    // x-path 3x3 "deconvs" (stride1 pad1) for reset/update/out gates
    deconv_kernel<<<nb(2LL * 64 * 128 * 128), TB, 0, stream>>>(x1, rg_w_1, rg_b_1, rxb,
                                                               2, 64, 64, 128, 128, 128, 128, 3, 1, 1);
    deconv_kernel<<<nb(2LL * 64 * 128 * 128), TB, 0, stream>>>(x1, ug_w_1, ug_b_1, uxb,
                                                               2, 64, 64, 128, 128, 128, 128, 3, 1, 1);
    deconv_kernel<<<nb(2LL * 64 * 128 * 128), TB, 0, stream>>>(x1, og_w_1, og_b_1, oxb,
                                                               2, 64, 64, 128, 128, 128, 128, 3, 1, 1);
    // waves = (64/16) * (32768/16) = 8192 -> 1024 blocks of 8 full waves
    gates_wmma_kernel<18, 14><<<1024, TB, 0, stream>>>(M1, Af1, rgw_b_1, ugw_b_1, ogw_b_1,
                                                       h1, rxb, uxb, oxb, s1,
                                                       64, /*PT=*/2048);
    // upsample: s1 (64,128,128) -> x2 (32,256,256)
    deconv_kernel<<<nb(2LL * 32 * 256 * 256), TB, 0, stream>>>(s1, up_w_1, up_b_1, x2,
                                                               2, 64, 32, 128, 128, 256, 256, 4, 2, 1);
    // final 1x1 projection -> (2,1,256,256)
    deconv_kernel<<<nb(2LL * 1 * 256 * 256), TB, 0, stream>>>(x2, out_w, out_b, (float*)d_out,
                                                              2, 32, 1, 256, 256, 256, 256, 1, 1, 0);
}
